// SelfAttnHead_49203145343519
// MI455X (gfx1250) — matrix operand order repaired and checked by reading, not compiled, not run on hardware
//
#include <hip/hip_runtime.h>
#include <hip/hip_bf16.h>

// MI455X / gfx1250 flash-attention-style single-head causal attention.
// All GEMMs run on v_wmma_f32_16x16x32_bf16 (bf16 in, f32 accumulate).
// All global traffic forced to address_space(1) -> global_load/global_store.

#define GAS __attribute__((address_space(1)))

typedef __attribute__((ext_vector_type(16))) __bf16 v16bf;
typedef __attribute__((ext_vector_type(8)))  __bf16 v8bf;
typedef __attribute__((ext_vector_type(8)))  float  v8f;
typedef __attribute__((ext_vector_type(4)))  float  f32x4;
typedef __attribute__((ext_vector_type(2)))  unsigned u32x2;

#define BATCH 4
#define SEQ   2048
#define DMODEL 1024
#define DHEAD 128
#define NEG_INF (-100000000.0f)
#define SCALE 0.08838834764831845f /* 1/sqrt(128) */

__device__ __forceinline__ unsigned short f2bf_us(float f) {
  union { float f; unsigned u; } v; v.f = f;
  unsigned r = v.u + 0x7FFFu + ((v.u >> 16) & 1u);   // round-to-nearest-even
  return (unsigned short)(r >> 16);
}

__device__ __forceinline__ v16bf cat8(v8bf lo, v8bf hi) {
  return __builtin_shufflevector(lo, hi, 0,1,2,3,4,5,6,7,8,9,10,11,12,13,14,15);
}

__device__ __forceinline__ v8f zero8() {
  v8f z = {0.f,0.f,0.f,0.f,0.f,0.f,0.f,0.f};
  return z;
}

// ---------------- conversion kernels ----------------

__global__ void cvt_x_kernel(const float* __restrict__ x, unsigned short* __restrict__ xb) {
  int i = (blockIdx.x * blockDim.x + threadIdx.x) * 4;
  f32x4 v = *(GAS const f32x4*)(x + i);
  u32x2 p;
  p.x = ((unsigned)f2bf_us(v.y) << 16) | f2bf_us(v.x);
  p.y = ((unsigned)f2bf_us(v.w) << 16) | f2bf_us(v.z);
  *(GAS u32x2*)(xb + i) = p;
}

// WT[n][k] = W[n][k], bf16.  W is [128 x 1024], WT is [128 x 1024].
__global__ void cvt_wT_kernel(const float* __restrict__ W, unsigned short* __restrict__ WT) {
  int i = blockIdx.x * blockDim.x + threadIdx.x;   // 0..131071
  int k = i >> 7, n = i & 127;
  ((GAS unsigned short*)WT)[n * DMODEL + k] = f2bf_us(((GAS const float*)W)[n * DMODEL + k]);
}

// ---------------- QKV projection (WMMA GEMM) ----------------
// Grid: 512 blocks (one per 16-row M-tile of the 8192-row x), 256 threads (8 waves).
// Each wave owns 3 of the 24 {matrix, n-chunk} 16x16 output tiles. No LDS, no
// barriers: A operand loaded directly from global in WMMA A-layout (L0-shared).
__global__ void proj_kernel(const unsigned short* __restrict__ Xb,
                            const unsigned short* __restrict__ WTq,
                            const unsigned short* __restrict__ WTk,
                            const unsigned short* __restrict__ WTv,
                            const float* __restrict__ bq,
                            const float* __restrict__ bk,
                            const float* __restrict__ bv,
                            unsigned short* __restrict__ Qb,
                            unsigned short* __restrict__ KTb,
                            unsigned short* __restrict__ Vb) {
  const int tid  = threadIdx.x;
  const int wave = tid >> 5;
  const int lane = tid & 31;
  const int lh   = lane & 15;
  const int hi   = lane >> 4;
  const int G0   = blockIdx.x * 16;          // global row base (0..8191)

  const unsigned short* WTsel[3] = { WTq, WTk, WTv };
  int tmat[3], tnc[3];
  GAS const unsigned short* wb[3];
#pragma unroll
  for (int i = 0; i < 3; ++i) {
    int t = wave * 3 + i;
    tmat[i] = t >> 3;
    tnc[i]  = t & 7;
    wb[i] = (GAS const unsigned short*)WTsel[tmat[i]] + (size_t)(tnc[i] * 16 + lh) * DMODEL + hi * 8;
  }

  v8f acc[3];
#pragma unroll
  for (int i = 0; i < 3; ++i) acc[i] = zero8();

  GAS const unsigned short* ap =
      (GAS const unsigned short*)Xb + (size_t)(G0 + lh) * DMODEL + hi * 8;

#pragma unroll 2
  for (int ks = 0; ks < DMODEL; ks += 32) {
    // A operand (16x32 bf16, WMMA A layout): row = lh, k-groups split by lane half
    v8bf alo = *(GAS const v8bf*)(ap + ks);
    v8bf ahi = *(GAS const v8bf*)(ap + ks + 16);
    v16bf A = cat8(alo, ahi);

#pragma unroll
    for (int i = 0; i < 3; ++i) {
      // B operand: lane = N column, k-groups split by lane half (same as A)
      v16bf Bv = cat8(*(GAS const v8bf*)(wb[i] + ks), *(GAS const v8bf*)(wb[i] + ks + 16));
      acc[i] = __builtin_amdgcn_wmma_f32_16x16x32_bf16(false, A, false, Bv,
                                                       (short)0, acc[i], false, false);
    }
  }

  const float* bias[3] = { bq, bk, bv };
#pragma unroll
  for (int i = 0; i < 3; ++i) {
    float bvv = ((GAS const float*)bias[tmat[i]])[tnc[i] * 16 + lh];
#pragma unroll
    for (int r = 0; r < 8; ++r) {
      int M = r + 8 * hi;
      int grow = G0 + M;
      int colg = tnc[i] * 16 + lh;
      unsigned short h = f2bf_us(acc[i][r] + bvv);
      if (tmat[i] == 0) {
        ((GAS unsigned short*)Qb)[(size_t)grow * DHEAD + colg] = h;
      } else if (tmat[i] == 1) {
        ((GAS unsigned short*)KTb)[(size_t)grow * DHEAD + colg] = h;
      } else { // V stored transposed per batch: VT[b][d][s]
        int bb = grow >> 11;
        int sr = grow & (SEQ - 1);
        ((GAS unsigned short*)Vb)[((size_t)(bb * DHEAD + colg)) * SEQ + sr] = h;
      }
    }
  }
}

// ---------------- flash attention (causal) ----------------
// One wave per 16-row Q tile. 512 tiles -> 64 blocks x 256 threads (8 waves).
__global__ void attn_kernel(const unsigned short* __restrict__ Qb,
                            const unsigned short* __restrict__ KTb,
                            const unsigned short* __restrict__ Vb,
                            float* __restrict__ out) {
  __shared__ __align__(32) unsigned short P[8][16 * 32];

  const int wave = threadIdx.x >> 5;
  const int lane = threadIdx.x & 31;
  const int lh   = lane & 15;
  const int hi   = lane >> 4;
  const int tile = blockIdx.x * 8 + wave;    // 0..511
  const int b    = tile >> 7;
  const int m0   = (tile & 127) * 16;        // row base within batch
  unsigned short* Pw = &P[wave][0];
  unsigned* Pw32 = (unsigned*)Pw;

  // PV product uses identity k order: A column k holds key j+k; vperm only offsets
  // the V prefetch addresses, not the V B-operand loads.
  const int vperm = (lane >> 1) + ((lane & 1) << 4);

  // preload Q tile (16x128) into 4 WMMA A registers
  v16bf qa[4];
#pragma unroll
  for (int kc = 0; kc < 4; ++kc) {
    GAS const unsigned short* qp =
        (GAS const unsigned short*)Qb + (size_t)(tile * 16 + lh) * DHEAD + kc * 32 + hi * 8;
    qa[kc] = cat8(*(GAS const v8bf*)qp, *(GAS const v8bf*)(qp + 16));
  }

  v8f o[8];
#pragma unroll
  for (int t = 0; t < 8; ++t) o[t] = zero8();
  float mrow[8], lrow[8];
#pragma unroll
  for (int r = 0; r < 8; ++r) { mrow[r] = -3.0e38f; lrow[r] = 0.f; }

  const int jend = m0 + 16;                  // causal: columns <= max row in tile
  for (int j = 0; j < jend; j += 32) {
    const bool more = (j + 32) < jend;
    // scores: two 16x16 tiles (columns j..j+15, j+16..j+31), chained over K=128
    v8f s0 = zero8(), s1 = zero8();
#pragma unroll
    for (int kc = 0; kc < 4; ++kc) {
      GAS const unsigned short* kp =
          (GAS const unsigned short*)KTb + ((size_t)(b * SEQ + j + lh)) * DHEAD + kc * 32 + hi * 8;
      v16bf b0 = cat8(*(GAS const v8bf*)kp, *(GAS const v8bf*)(kp + 16));
      v16bf b1 = cat8(*(GAS const v8bf*)(kp + 16 * DHEAD), *(GAS const v8bf*)(kp + 16 * DHEAD + 16));
      if (more) __builtin_prefetch(KTb + ((size_t)(b * DHEAD + kc * 32 + lane)) * SEQ + j + 32, 0, 1);
      s0 = __builtin_amdgcn_wmma_f32_16x16x32_bf16(false, qa[kc], false, b0,
                                                   (short)0, s0, false, false);
      s1 = __builtin_amdgcn_wmma_f32_16x16x32_bf16(false, qa[kc], false, b1,
                                                   (short)0, s1, false, false);
    }

    // online softmax update (per-row stats; rows of C tile live in half-waves)
#pragma unroll
    for (int r = 0; r < 8; ++r) {
      int gi = m0 + r + 8 * hi;
      float a0 = s0[r] * SCALE;
      float a1 = s1[r] * SCALE;
      a0 = ((j + lh) > gi || a0 == 0.f) ? NEG_INF : a0;        // tril + ==0 quirk
      a1 = ((j + 16 + lh) > gi || a1 == 0.f) ? NEG_INF : a1;
      float t = fmaxf(a0, a1);
      t = fmaxf(t, __shfl_xor(t, 1));
      t = fmaxf(t, __shfl_xor(t, 2));
      t = fmaxf(t, __shfl_xor(t, 4));
      t = fmaxf(t, __shfl_xor(t, 8));
      float nm = fmaxf(mrow[r], t);
      float alpha = __expf(mrow[r] - nm);
      float p0 = __expf(a0 - nm);
      float p1 = __expf(a1 - nm);
      float rs = p0 + p1;
      rs += __shfl_xor(rs, 1);
      rs += __shfl_xor(rs, 2);
      rs += __shfl_xor(rs, 4);
      rs += __shfl_xor(rs, 8);
      lrow[r] = lrow[r] * alpha + rs;
      mrow[r] = nm;
#pragma unroll
      for (int t2 = 0; t2 < 8; ++t2) o[t2][r] *= alpha;
      // store: k positions lh (p0) and 16+lh (p1), row M
      int M = r + 8 * hi;
      Pw[M * 32 + lh] = f2bf_us(p0); Pw[M * 32 + 16 + lh] = f2bf_us(p1);
    }

    // reload P in WMMA A layout (16x32 bf16), same-wave LDS so no barrier needed
    v8bf plo = *(const v8bf*)&Pw[lh * 32 + hi * 8];
    v8bf phi = *(const v8bf*)&Pw[lh * 32 + 16 + hi * 8];
    v16bf pa = cat8(plo, phi);

    const size_t vrow = (size_t)(b * SEQ + j + vperm) * DHEAD;
    if (more) {
      __builtin_prefetch(Vb + vrow + 32 * DHEAD, 0, 1);
      __builtin_prefetch(Vb + vrow + 32 * DHEAD + 64, 0, 1);
    }
#pragma unroll
    for (int t2 = 0; t2 < 8; ++t2) {
      // B operand: lane = d column (VT row), k-groups split by lane half
      GAS const unsigned short* vp = (GAS const unsigned short*)Vb + ((size_t)(b * DHEAD + t2 * 16 + lh)) * SEQ + j + hi * 8;
      v16bf vb = cat8(*(GAS const v8bf*)vp, *(GAS const v8bf*)(vp + 16));
      o[t2] = __builtin_amdgcn_wmma_f32_16x16x32_bf16(false, pa, false, vb,
                                                      (short)0, o[t2], false, false);
    }
  }

  // normalize and store fp32 output
  float inv[8];
#pragma unroll
  for (int r = 0; r < 8; ++r) inv[r] = 1.0f / lrow[r];
#pragma unroll
  for (int t2 = 0; t2 < 8; ++t2) {
#pragma unroll
    for (int r = 0; r < 8; ++r) {
      size_t idx = (size_t)(tile * 16 + r + 8 * hi) * DHEAD + t2 * 16 + lh;
      ((GAS float*)out)[idx] = o[t2][r] * inv[r];
    }
  }
}

// ---------------- launcher ----------------

extern "C" void kernel_launch(void* const* d_in, const int* in_sizes, int n_in,
                              void* d_out, int out_size, void* d_ws, size_t ws_size,
                              hipStream_t stream) {
  (void)in_sizes; (void)n_in; (void)out_size; (void)ws_size;

  const float* x  = (const float*)d_in[0];
  const float* Wq = (const float*)d_in[1];
  const float* bq = (const float*)d_in[2];
  const float* Wk = (const float*)d_in[3];
  const float* bk = (const float*)d_in[4];
  const float* Wv = (const float*)d_in[5];
  const float* bv = (const float*)d_in[6];
  float* out = (float*)d_out;

  char* ws = (char*)d_ws;
  // workspace layout (bytes), all 32B-aligned
  unsigned short* Xb  = (unsigned short*)(ws);                       // 8192*1024 bf16 = 16 MB
  unsigned short* WTq = (unsigned short*)(ws + 16777216);            // 1024*128
  unsigned short* WTk = (unsigned short*)(ws + 16777216 + 262144);
  unsigned short* WTv = (unsigned short*)(ws + 16777216 + 524288);
  unsigned short* Qb  = (unsigned short*)(ws + 17563648);            // 8192*128
  unsigned short* KTb = (unsigned short*)(ws + 17563648 + 2097152);  // 4*128*2048
  unsigned short* Vb  = (unsigned short*)(ws + 17563648 + 4194304);  // 8192*128

  const int nx = BATCH * SEQ * DMODEL;   // 8388608
  cvt_x_kernel<<<nx / 4 / 256, 256, 0, stream>>>(x, Xb);
  cvt_wT_kernel<<<(DMODEL * DHEAD) / 256, 256, 0, stream>>>(Wq, WTq);
  cvt_wT_kernel<<<(DMODEL * DHEAD) / 256, 256, 0, stream>>>(Wk, WTk);
  cvt_wT_kernel<<<(DMODEL * DHEAD) / 256, 256, 0, stream>>>(Wv, WTv);

  proj_kernel<<<(BATCH * SEQ) / 16, 256, 0, stream>>>(Xb, WTq, WTk, WTv,
                                                      bq, bk, bv, Qb, KTb, Vb);

  attn_kernel<<<(BATCH * SEQ) / 16 / 8, 256, 0, stream>>>(Qb, KTb, Vb, out);
}
